// FractalAttention_30709016167035
// MI455X (gfx1250) — matrix operand order repaired and checked by reading, not compiled, not run on hardware
//
#include <hip/hip_runtime.h>
#include <hip/hip_bf16.h>
#include <math.h>

// ---------------------------------------------------------------------------
// FractalAttention on MI455X (gfx1250), wave32 + v_wmma_f32_16x16x32_bf16.
// B=4, T=1024, D=512, H=8, hd=64; 5 attention levels, device-side gating.
// Register-blocked WMMA (32x64 GEMM tiles / 4-wide attention blocks) so each
// fragment load feeds multiple WMMAs; all fragments are b128 vector loads.
// ---------------------------------------------------------------------------

typedef __attribute__((ext_vector_type(16))) __bf16 v16bf;
typedef __attribute__((ext_vector_type(8)))  __bf16 v8bf;
typedef __attribute__((ext_vector_type(8)))  float  v8f;

#define B_    4
#define T_    1024
#define D_    512
#define H_    8
#define HD_   64
#define QKVN  1536          // 3*D
#define ATT_WAVES 2         // waves per attention block

__device__ __forceinline__ v16bf load_pair(const __bf16* p0, const __bf16* p1) {
    v8bf lo = *(const v8bf*)p0;
    v8bf hi = *(const v8bf*)p1;
    return __builtin_shufflevector(lo, hi, 0, 1, 2, 3, 4, 5, 6, 7,
                                           8, 9, 10, 11, 12, 13, 14, 15);
}

__device__ __forceinline__ v8f wmma_bf16(v16bf a, v16bf b, v8f c) {
    return __builtin_amdgcn_wmma_f32_16x16x32_bf16(
        false, a, false, b, (short)0, c, false, false);
}

// ---------------- utility kernels ------------------------------------------

__global__ void zero_f32_kernel(float* __restrict__ p, int n) {
    int i = blockIdx.x * blockDim.x + threadIdx.x;
    if (i < n) p[i] = 0.0f;
}

__global__ void cvt_f32_bf16_kernel(const float* __restrict__ in,
                                    __bf16* __restrict__ out, int n) {
    int i = blockIdx.x * blockDim.x + threadIdx.x;
    int stride = gridDim.x * blockDim.x;
    for (; i < n; i += stride) out[i] = (__bf16)in[i];
}

// in: [K,N] f32 row-major  ->  out: [N,K] bf16 row-major (transposed)
__global__ void cvt_transpose_kernel(const float* __restrict__ in,
                                     __bf16* __restrict__ out, int K, int N) {
    int idx = blockIdx.x * blockDim.x + threadIdx.x;
    int stride = gridDim.x * blockDim.x;
    const int total = K * N;
    for (; idx < total; idx += stride) {
        int k = idx % K;
        int n = idx / K;
        out[idx] = (__bf16)in[(size_t)k * N + n];
    }
}

// vT[((b*H+h)*HD+d)*T + t] = qkv[(b*T+t)*QKVN + 2D + h*HD + d]
__global__ void build_vT_kernel(const __bf16* __restrict__ qkv,
                                __bf16* __restrict__ vT) {
    int idx = blockIdx.x * blockDim.x + threadIdx.x;
    int stride = gridDim.x * blockDim.x;
    const int total = B_ * H_ * HD_ * T_;
    for (; idx < total; idx += stride) {
        int t = idx & (T_ - 1);
        int d = (idx >> 10) & (HD_ - 1);
        int h = (idx >> 16) & (H_ - 1);
        int b = idx >> 19;
        vT[idx] = qkv[(size_t)(b * T_ + t) * QKVN + 2 * D_ + h * HD_ + d];
    }
}

// ---------------- bf16 WMMA GEMM:  C[M,N] = A[M,K] @ Bt^T + bias -----------
// A: [M,K] bf16 row-major.  Bt: [N,K] bf16 row-major (pre-transposed B).
// One wave -> 32(M) x 64(N) register-blocked macro-tile (2x4 WMMA tiles):
// per K-step of 32, 2 A-frags + 4 B-frags feed 8 WMMAs.
// Fragment layouts used here (wave32, contiguous k order, not the ISA table order):
//   A: row = lane&15; elem i -> K = kk + hi*16 + i
//   B: col = lane&15; elem i -> K = kk + hi*16 + i   (contiguous in Bt)
//   C: col = lane&15; vgpr i -> row hi*8 + i

__global__ void gemm_bf16_kernel(const __bf16* __restrict__ A,
                                 const __bf16* __restrict__ Bt,
                                 const float*  __restrict__ bias,
                                 void* __restrict__ Cout,
                                 int M, int N, int K, int out_is_bf16) {
    const int lane = threadIdx.x & 31;
    const int wave = threadIdx.x >> 5;
    const int tile = blockIdx.x * (blockDim.x >> 5) + wave;
    const int tilesN = N >> 6;                 // 64-wide macro-tiles
    const int tm = (tile / tilesN) << 5;       // 32-tall
    const int tn = (tile % tilesN) << 6;
    if (tm >= M) return;                       // wave-uniform
    const int r  = lane & 15;
    const int hi = lane >> 4;

    const __bf16* Ap0 = A + (size_t)(tm + r) * K + hi * 16;
    const __bf16* Ap1 = Ap0 + (size_t)16 * K;
    const __bf16* Bp0 = Bt + (size_t)(tn + r) * K + hi * 16;
    const __bf16* Bp1 = Bp0 + (size_t)16 * K;
    const __bf16* Bp2 = Bp0 + (size_t)32 * K;
    const __bf16* Bp3 = Bp0 + (size_t)48 * K;

    v8f acc[2][4] = {};
    for (int kk = 0; kk < K; kk += 32) {
        v16bf a0 = load_pair(Ap0 + kk, Ap0 + kk + 8);
        v16bf a1 = load_pair(Ap1 + kk, Ap1 + kk + 8);
        v16bf b0 = load_pair(Bp0 + kk, Bp0 + kk + 8);
        v16bf b1 = load_pair(Bp1 + kk, Bp1 + kk + 8);
        v16bf b2 = load_pair(Bp2 + kk, Bp2 + kk + 8);
        v16bf b3 = load_pair(Bp3 + kk, Bp3 + kk + 8);
        acc[0][0] = wmma_bf16(a0, b0, acc[0][0]);
        acc[1][0] = wmma_bf16(a1, b0, acc[1][0]);
        acc[0][1] = wmma_bf16(a0, b1, acc[0][1]);
        acc[1][1] = wmma_bf16(a1, b1, acc[1][1]);
        acc[0][2] = wmma_bf16(a0, b2, acc[0][2]);
        acc[1][2] = wmma_bf16(a1, b2, acc[1][2]);
        acc[0][3] = wmma_bf16(a0, b3, acc[0][3]);
        acc[1][3] = wmma_bf16(a1, b3, acc[1][3]);
    }

    if (out_is_bf16) {
        __bf16* C = (__bf16*)Cout;
#pragma unroll
        for (int mi = 0; mi < 2; ++mi)
#pragma unroll
            for (int ni = 0; ni < 4; ++ni)
#pragma unroll
                for (int i = 0; i < 8; ++i) {
                    int m = tm + mi * 16 + hi * 8 + i;
                    int n = tn + ni * 16 + r;
                    C[(size_t)m * N + n] = (__bf16)(acc[mi][ni][i] + bias[n]);
                }
    } else {
        float* C = (float*)Cout;
#pragma unroll
        for (int mi = 0; mi < 2; ++mi)
#pragma unroll
            for (int ni = 0; ni < 4; ++ni)
#pragma unroll
                for (int i = 0; i < 8; ++i) {
                    int m = tm + mi * 16 + hi * 8 + i;
                    int n = tn + ni * 16 + r;
                    C[(size_t)m * N + n] = acc[mi][ni][i] + bias[n];
                }
    }
}

// ---------------- fused attention level -------------------------------------
// qkv: bf16 [B*T, 1536]; q cols [0,512), k [512,1024).  vT: bf16 [B,H,hd,T].
// One wave owns a 16-row query strip of one (b,h):
//   scores: q A-frags hoisted (loop-invariant), 4 key-tiles blocked per pass
//   softmax + entropy (wave32 shfl reductions) -> bf16 probs in LDS
//   attn @ v: 4 N-accumulators live across K loop; one LDS A-frag / 4 WMMAs

__global__ void fractal_attn_kernel(const __bf16* __restrict__ qkv,
                                    const __bf16* __restrict__ vT,
                                    float* __restrict__ out,
                                    float* __restrict__ ent_acc,
                                    float scale) {
    extern __shared__ char smem_raw[];
    float*  Sall = (float*)smem_raw;                                // scores f32
    __bf16* Pall = (__bf16*)(smem_raw + ATT_WAVES * 16 * T_ * 4);   // probs bf16

    const int lane  = threadIdx.x & 31;
    const int wave  = threadIdx.x >> 5;
    const int strip = blockIdx.x * ATT_WAVES + wave;
    const int stripsPerBH = T_ / 16;           // 64
    const int bh = strip / stripsPerBH;
    const int t0 = (strip % stripsPerBH) * 16;
    const int bb = bh / H_;
    const int hh = bh % H_;
    const int r  = lane & 15;
    const int hi = lane >> 4;

    float*  S = Sall + (size_t)wave * 16 * T_;
    __bf16* P = Pall + (size_t)wave * 16 * T_;

    const size_t ld = QKVN;
    const __bf16* Qb  = qkv + (size_t)bb * T_ * ld + hh * HD_;
    const __bf16* Kb  = Qb + D_;
    const __bf16* VTb = vT + (size_t)(bb * H_ + hh) * HD_ * T_;

    // ---- scores: S[16 x T] = scale * (q @ k^T); q frags loop-invariant
    const __bf16* Aq = Qb + (size_t)(t0 + r) * ld + hi * 16;
    const v16bf aq0 = load_pair(Aq, Aq + 8);         // K = 0..31 slice of hd
    const v16bf aq1 = load_pair(Aq + 32, Aq + 40);   // K = 32..63 slice
    for (int jt = 0; jt < T_ / 16; jt += 4) {
        v8f acc[4] = {};
#pragma unroll
        for (int j = 0; j < 4; ++j) {
            const __bf16* Bk = Kb + (size_t)((jt + j) * 16 + r) * ld + hi * 16;
            acc[j] = wmma_bf16(aq0, load_pair(Bk, Bk + 8), acc[j]);
            acc[j] = wmma_bf16(aq1, load_pair(Bk + 32, Bk + 40), acc[j]);
        }
#pragma unroll
        for (int j = 0; j < 4; ++j)
#pragma unroll
            for (int i = 0; i < 8; ++i)
                S[(hi * 8 + i) * T_ + (jt + j) * 16 + r] = acc[j][i] * scale;
    }
    __syncthreads();

    // ---- softmax over each row (f32) -> bf16 probs + entropy
    float ent_local = 0.0f;
    for (int row = 0; row < 16; ++row) {
        float*  Sr = S + row * T_;
        __bf16* Pr = P + row * T_;
        float m = -3.4e38f;
        for (int c = lane; c < T_; c += 32) m = fmaxf(m, Sr[c]);
#pragma unroll
        for (int off = 16; off >= 1; off >>= 1)
            m = fmaxf(m, __shfl_xor(m, off, 32));
        float sum = 0.0f;
        for (int c = lane; c < T_; c += 32) sum += __expf(Sr[c] - m);
#pragma unroll
        for (int off = 16; off >= 1; off >>= 1)
            sum += __shfl_xor(sum, off, 32);
        const float inv  = 1.0f / sum;
        const float lsum = __logf(sum);
        float plogp = 0.0f;
        for (int c = lane; c < T_; c += 32) {
            float e = Sr[c] - m;
            float p = __expf(e) * inv;
            Pr[c] = (__bf16)p;
            plogp += p * (e - lsum);           // p * log(p)
        }
#pragma unroll
        for (int off = 16; off >= 1; off >>= 1)
            plogp += __shfl_xor(plogp, off, 32);
        ent_local += -plogp;
    }
    if (lane == 0) atomicAdd(ent_acc, ent_local);
    __syncthreads();

    // ---- out = attn @ v : 4 N-accumulators, one LDS A-frag per K-step
    const __bf16* Arow = P + (size_t)r * T_ + hi * 16;
    const __bf16* Bv0 = VTb + (size_t)r * T_ + hi * 16;
    const __bf16* Bv1 = Bv0 + (size_t)16 * T_;
    const __bf16* Bv2 = Bv0 + (size_t)32 * T_;
    const __bf16* Bv3 = Bv0 + (size_t)48 * T_;
    v8f acc[4] = {};
    for (int kk = 0; kk < T_; kk += 32) {
        v16bf a = load_pair(Arow + kk, Arow + kk + 8);    // ds_load_b128 x2
        acc[0] = wmma_bf16(a, load_pair(Bv0 + kk, Bv0 + kk + 8), acc[0]);
        acc[1] = wmma_bf16(a, load_pair(Bv1 + kk, Bv1 + kk + 8), acc[1]);
        acc[2] = wmma_bf16(a, load_pair(Bv2 + kk, Bv2 + kk + 8), acc[2]);
        acc[3] = wmma_bf16(a, load_pair(Bv3 + kk, Bv3 + kk + 8), acc[3]);
    }
#pragma unroll
    for (int nt = 0; nt < 4; ++nt)
#pragma unroll
        for (int i = 0; i < 8; ++i) {
            int t = t0 + hi * 8 + i;
            out[((size_t)bb * T_ + t) * D_ + hh * HD_ + nt * 16 + r] = acc[nt][i];
        }
}

// ---------------- gate + combine --------------------------------------------

__global__ void gate_kernel(const float* __restrict__ ent,
                            float* __restrict__ coeffs) {
    if (threadIdx.x == 0 && blockIdx.x == 0) {
        const float denom = (float)(B_ * H_ * T_) * __logf((float)T_);
        float c = 1.0f;
        coeffs[0] = 1.0f;
        for (int l = 0; l < 4; ++l) {
            float e = ent[l] / denom;
            c *= (e < 0.8f) ? 1.0f : 0.0f;
            coeffs[l + 1] = c;
        }
    }
}

__global__ void combine_kernel(const float* __restrict__ o0,
                               const float* __restrict__ o1,
                               const float* __restrict__ o2,
                               const float* __restrict__ o3,
                               const float* __restrict__ o4,
                               const float* __restrict__ coeffs,
                               __bf16* __restrict__ comb, int n) {
    const float c1 = coeffs[1], c2 = coeffs[2], c3 = coeffs[3], c4 = coeffs[4];
    int i = blockIdx.x * blockDim.x + threadIdx.x;
    int stride = gridDim.x * blockDim.x;
    for (; i < n; i += stride)
        comb[i] = (__bf16)(o0[i] + c1 * o1[i] + c2 * o2[i] + c3 * o3[i] + c4 * o4[i]);
}

// ---------------- host orchestration ----------------------------------------

extern "C" void kernel_launch(void* const* d_in, const int* in_sizes, int n_in,
                              void* d_out, int out_size, void* d_ws, size_t ws_size,
                              hipStream_t stream) {
    (void)in_sizes; (void)n_in; (void)out_size; (void)ws_size;
    const float* x    = (const float*)d_in[0];
    const float* wqkv = (const float*)d_in[1];
    const float* bqkv = (const float*)d_in[2];
    const float* wout = (const float*)d_in[3];
    const float* bout = (const float*)d_in[4];

    char* ws = (char*)d_ws;
    size_t off = 0;
    auto alloc = [&](size_t bytes) -> void* {
        void* p = ws + off;
        off = (off + bytes + 255) & ~(size_t)255;
        return p;
    };
    float*  ent     = (float*)alloc(8 * sizeof(float));
    float*  coeffs  = (float*)alloc(8 * sizeof(float));
    __bf16* wqkvT_b = (__bf16*)alloc((size_t)D_ * QKVN * 2);     // [1536, 512]
    __bf16* woutT_b = (__bf16*)alloc((size_t)D_ * D_ * 2);       // [512, 512]
    __bf16* x_b     = (__bf16*)alloc((size_t)B_ * T_ * D_ * 2);
    __bf16* qkv_b   = (__bf16*)alloc((size_t)B_ * T_ * QKVN * 2);
    __bf16* vT_b    = (__bf16*)alloc((size_t)B_ * T_ * D_ * 2);  // [B,H,hd,T]
    float*  outs[5];
    for (int l = 0; l < 5; ++l)
        outs[l] = (float*)alloc((size_t)B_ * T_ * D_ * sizeof(float));
    __bf16* comb_b  = (__bf16*)alloc((size_t)B_ * T_ * D_ * 2);

    const int nAct = B_ * T_ * D_;
    zero_f32_kernel<<<1, 32, 0, stream>>>(ent, 8);
    cvt_transpose_kernel<<<512, 256, 0, stream>>>(wqkv, wqkvT_b, D_, QKVN);
    cvt_transpose_kernel<<<256, 256, 0, stream>>>(wout, woutT_b, D_, D_);
    cvt_f32_bf16_kernel<<<512, 256, 0, stream>>>(x, x_b, nAct);

    const float scale = 1.0f / sqrtf((float)D_);   // embed_dim scaling (matches ref)

    const int gemmWavesPerBlock = 4;
    const int gemmBlock = 32 * gemmWavesPerBlock;
    // 32x64 macro-tiles
    const int qkvTiles = (B_ * T_ / 32) * (QKVN / 64);       // 128*24 = 3072
    const int qkvGrid  = qkvTiles / gemmWavesPerBlock;       // 768
    const int attnStrips = B_ * H_ * (T_ / 16);              // 2048
    const int attnGrid   = attnStrips / ATT_WAVES;           // 1024
    const size_t attnLds = (size_t)ATT_WAVES * 16 * T_ * (4 + 2); // 192 KB

    for (int l = 0; l < 5; ++l) {
        gemm_bf16_kernel<<<qkvGrid, gemmBlock, 0, stream>>>(
            x_b, wqkvT_b, bqkv, qkv_b, B_ * T_, QKVN, D_, 1);
        build_vT_kernel<<<512, 256, 0, stream>>>(qkv_b, vT_b);
        fractal_attn_kernel<<<attnGrid, ATT_WAVES * 32, attnLds, stream>>>(
            qkv_b, vT_b, outs[l], ent + l, scale);
        if (l < 4)
            cvt_f32_bf16_kernel<<<512, 256, 0, stream>>>(outs[l], x_b, nAct);
    }

    gate_kernel<<<1, 32, 0, stream>>>(ent, coeffs);
    combine_kernel<<<512, 256, 0, stream>>>(
        outs[0], outs[1], outs[2], outs[3], outs[4], coeffs, comb_b, nAct);

    const int outTiles = (B_ * T_ / 32) * (D_ / 64);         // 128*8 = 1024
    gemm_bf16_kernel<<<outTiles / gemmWavesPerBlock, gemmBlock, 0, stream>>>(
        comb_b, woutT_b, bout, (float*)d_out, B_ * T_, D_, D_, 0);
}
